// CrystalGCN_38938173506038
// MI455X (gfx1250) — compile-verified, matrix operand order checked by reading, not run on hardware
//
#include <hip/hip_runtime.h>

// ---------------- problem constants (from reference) ----------------
#define N_NODES   50000
#define N_EDGES   800000
#define HIDDEN    128
#define NUM_RBF   32
#define N_GRAPHS  500
#define ZDIM      288             // 2*HIDDEN + NUM_RBF
#define ZPAD      296             // padded LDS row stride (elems): 16B-aligned, spreads banks
#define NGROUPS   (N_EDGES / 128) // 6250 groups of 128 edges (8 A-tiles), exact

typedef __attribute__((ext_vector_type(16))) __bf16 v16bf;
typedef __attribute__((ext_vector_type(8)))  float  v8f;

union Frag { v16bf v; uint4 q[2]; };
union Acc  { v8f v; float f[8]; };

__device__ __forceinline__ unsigned short f2bf(float f) {
  union { float f; unsigned int u; } v; v.f = f;
  unsigned int u = v.u;
  u += 0x7FFFu + ((u >> 16) & 1u);       // round-to-nearest-even to bf16
  return (unsigned short)(u >> 16);
}

__device__ __forceinline__ v16bf load_frag(const char* base, int off) {
  Frag r;
  r.q[0] = *(const uint4*)(base + off);       // K 0..7   (or 8..15 for hi lanes)
  r.q[1] = *(const uint4*)(base + off + 32);  // K 16..23 (or 24..31 for hi lanes)
  return r.v;
}

// -------- weight prep: 6x [288][128] f32 -> transposed [128][288] bf16 --------
__global__ void wconv_kernel(const float* __restrict__ W1f, const float* __restrict__ W1s,
                             const float* __restrict__ W2f, const float* __restrict__ W2s,
                             const float* __restrict__ W3f, const float* __restrict__ W3s,
                             unsigned short* __restrict__ out) {
  int tid = blockIdx.x * blockDim.x + threadIdx.x;
  const int per = ZDIM * HIDDEN;
  if (tid >= 6 * per) return;
  int m = tid / per, rem = tid % per;
  int k = rem / HIDDEN, n = rem % HIDDEN;
  const float* W = (m == 0) ? W1f : (m == 1) ? W1s : (m == 2) ? W2f
                 : (m == 3) ? W2s : (m == 4) ? W3f : W3s;
  out[(size_t)m * per + (size_t)n * ZDIM + k] = f2bf(W[(size_t)k * HIDDEN + n]);
}

// -------- edge_attr f32 -> bf16 (once per launch, reused by all 3 layers) --------
__global__ void eaconv_kernel(const float* __restrict__ ea, unsigned short* __restrict__ eab) {
  int tid = blockIdx.x * blockDim.x + threadIdx.x;
  if (tid < N_EDGES * NUM_RBF) eab[tid] = f2bf(ea[tid]);
}

__global__ void embed_kernel(const int* __restrict__ x, const float* __restrict__ emb,
                             float* __restrict__ h, unsigned short* __restrict__ hb) {
  int tid = blockIdx.x * blockDim.x + threadIdx.x;
  if (tid >= N_NODES * HIDDEN) return;
  int n = tid >> 7, c = tid & 127;
  float v = emb[(size_t)x[n] * HIDDEN + c];
  h[tid] = v;
  hb[tid] = f2bf(v);
}

__global__ void copy4_kernel(const float4* __restrict__ s, float4* __restrict__ d, int n4) {
  int tid = blockIdx.x * blockDim.x + threadIdx.x;
  if (tid < n4) d[tid] = s[tid];
}

// h = relu(agg), plus bf16 mirror for next layer's WMMA staging
__global__ void relu_kernel(const float* __restrict__ s, float* __restrict__ d,
                            unsigned short* __restrict__ db, int n) {
  int tid = blockIdx.x * blockDim.x + threadIdx.x;
  if (tid < n) {
    float v = fmaxf(s[tid], 0.0f);
    d[tid] = v;
    db[tid] = f2bf(v);
  }
}

// -------- per-layer edge-message kernel: WMMA bf16 GEMMs + gated scatter --------
// Block stages 128 edges (8 A-tiles of 16) into LDS; wave w computes A-tile pair
// (w&3)*2,(w&3)*2+1 for n-columns [(w>>2)*64, +64), for both gate (F) and soft (S)
// branches. Each B fragment (weights, LDS-resident) feeds two back-to-back WMMAs.
// Next group's gathers are prefetched (global_prefetch_b8) while WMMAs run.
__global__ void __launch_bounds__(256)
cg_msg_kernel(const unsigned short* __restrict__ hb, const int* __restrict__ esrc,
              const int* __restrict__ edst, const unsigned short* __restrict__ eab,
              const unsigned short* __restrict__ wfT, const unsigned short* __restrict__ wsT,
              const float* __restrict__ biasF, const float* __restrict__ biasS,
              float* __restrict__ agg) {
  extern __shared__ unsigned short smem[];
  unsigned short* wfL  = smem;                       // [128][296] bf16 W_f^T
  unsigned short* wsL  = smem + HIDDEN * ZPAD;       // [128][296] bf16 W_s^T
  unsigned short* zAll = smem + 2 * HIDDEN * ZPAD;   // [128 rows][296] staged z (bf16)
  int* dstAll = (int*)(zAll + 128 * ZPAD);           // 128 dst node indices

  // cooperative weight load: 256 threads, one 288-elem row each (rows are N of W^T)
  {
    int t = threadIdx.x;
    const unsigned short* g = (t < HIDDEN) ? (wfT + (size_t)t * ZDIM)
                                           : (wsT + (size_t)(t - HIDDEN) * ZDIM);
    unsigned short* d = (t < HIDDEN) ? (wfL + t * ZPAD) : (wsL + (t - HIDDEN) * ZPAD);
    const uint4* gs = (const uint4*)g;
    uint4* dd = (uint4*)d;
    #pragma unroll
    for (int i = 0; i < ZDIM / 8; ++i) dd[i] = gs[i];
  }

  const int lane = threadIdx.x & 31;
  const int wave = threadIdx.x >> 5;
  const int colsel = lane & 15;        // A: row within tile, B: col within n-tile
  const int hi = lane >> 4;            // selects K 8-element chunk (16-bit layout)
  const int aT0 = (wave & 3) * 2;      // first A-tile of this wave's pair
  const int nbase = (wave >> 2) * 4;   // first n-tile (4 per wave = 64 cols)

  // loop-invariant fragment byte offsets and biases
  const int aOff0 = (aT0 * 16 + colsel) * (ZPAD * 2) + hi * 16;
  const int aOff1 = aOff0 + 16 * (ZPAD * 2);
  int bOff[4];
  float bfv[4], bsv[4];
  #pragma unroll
  for (int nn = 0; nn < 4; ++nn) {
    int n = (nbase + nn) * 16 + colsel;
    bOff[nn] = n * (ZPAD * 2) + hi * 16;
    bfv[nn] = biasF[n];
    bsv[nn] = biasS[n];
  }

  const int r = threadIdx.x >> 1, half = threadIdx.x & 1;  // staging: 2 threads/row
  const int stride = gridDim.x;

  __syncthreads();  // weights resident

  for (int g = blockIdx.x; g < NGROUPS; g += stride) {
    // ---- stage 128 z-rows = [h[dst] | h[src] | e_ij] (already bf16, no cvt) ----
    const int e = g * 128 + r;
    const int srcI = esrc[e], dstI = edst[e];
    if (half == 0) dstAll[r] = dstI;
    {
      uint4* zr = (uint4*)(zAll + r * ZPAD);           // 16B chunks, chunk idx = elem/8
      const uint4* hd = (const uint4*)(hb + (size_t)dstI * HIDDEN);
      const uint4* hs = (const uint4*)(hb + (size_t)srcI * HIDDEN);
      const uint4* eb = (const uint4*)(eab + (size_t)e * NUM_RBF);
      if (half == 0) {      // elems [0,144): h_dst[0..127], h_src[0..15]
        #pragma unroll
        for (int i = 0; i < 16; ++i) zr[i] = hd[i];
        zr[16] = hs[0];
        zr[17] = hs[1];
      } else {              // elems [144,288): h_src[16..127], e[0..31]
        #pragma unroll
        for (int i = 0; i < 14; ++i) zr[18 + i] = hs[2 + i];
        #pragma unroll
        for (int i = 0; i < 4; ++i) zr[32 + i] = eb[i];
      }
    }
    // pull next group's edge indices while staging stores drain (covered by barrier)
    const int gn = g + stride;
    int srcN = 0, dstN = 0;
    if (gn < NGROUPS) {
      srcN = esrc[gn * 128 + r];
      dstN = edst[gn * 128 + r];
    }
    __syncthreads();  // staged z visible to all waves

    // prefetch next group's gather targets into cache; overlaps the WMMA sequence
    if (gn < NGROUPS) {
      if (half == 0) {
        __builtin_prefetch(hb + (size_t)dstN * HIDDEN, 0, 3);
        __builtin_prefetch(hb + (size_t)dstN * HIDDEN + 64, 0, 3);
        __builtin_prefetch(hb + (size_t)srcN * HIDDEN, 0, 3);
        __builtin_prefetch(hb + (size_t)srcN * HIDDEN + 64, 0, 3);
      } else {
        __builtin_prefetch(eab + (size_t)(gn * 128 + r) * NUM_RBF, 0, 3);
      }
    }

    int drow0[8], drow1[8];
    #pragma unroll
    for (int i = 0; i < 8; ++i) {
      drow0[i] = dstAll[aT0 * 16 + i + 8 * hi];        // M = vgpr + 8*hi
      drow1[i] = dstAll[aT0 * 16 + 16 + i + 8 * hi];
    }

    Acc acc[2][4][2];  // [A-tile][n-tile][F/S]  -> 128 f32 accumulator VGPRs
    #pragma unroll
    for (int a = 0; a < 2; ++a)
      #pragma unroll
      for (int nn = 0; nn < 4; ++nn)
        #pragma unroll
        for (int i = 0; i < 8; ++i) { acc[a][nn][0].f[i] = 0.0f; acc[a][nn][1].f[i] = 0.0f; }

    const char* zb  = (const char*)zAll;
    const char* wfb = (const char*)wfL;
    const char* wsb = (const char*)wsL;
    for (int kb = 0; kb < 9; ++kb) {          // K = 288 = 9 x 32
      v16bf a0 = load_frag(zb, aOff0 + kb * 64);
      v16bf a1 = load_frag(zb, aOff1 + kb * 64);
      #pragma unroll
      for (int nn = 0; nn < 4; ++nn) {
        int bo = bOff[nn] + kb * 64;
        v16bf bF = load_frag(wfb, bo);        // one LDS fragment -> two WMMAs
        acc[0][nn][0].v = __builtin_amdgcn_wmma_f32_16x16x32_bf16(
            false, a0, false, bF, (short)0, acc[0][nn][0].v, false, false);
        acc[1][nn][0].v = __builtin_amdgcn_wmma_f32_16x16x32_bf16(
            false, a1, false, bF, (short)0, acc[1][nn][0].v, false, false);
        v16bf bS = load_frag(wsb, bo);
        acc[0][nn][1].v = __builtin_amdgcn_wmma_f32_16x16x32_bf16(
            false, a0, false, bS, (short)0, acc[0][nn][1].v, false, false);
        acc[1][nn][1].v = __builtin_amdgcn_wmma_f32_16x16x32_bf16(
            false, a1, false, bS, (short)0, acc[1][nn][1].v, false, false);
      }
    }

    // ---- epilogue: msg = sigmoid(f+bf) * softplus(s+bs); scatter-add to agg[dst] ----
    #pragma unroll
    for (int nn = 0; nn < 4; ++nn) {
      int n = (nbase + nn) * 16 + colsel;
      #pragma unroll
      for (int a = 0; a < 2; ++a) {
        #pragma unroll
        for (int i = 0; i < 8; ++i) {
          float xf = acc[a][nn][0].f[i] + bfv[nn];
          float xs = acc[a][nn][1].f[i] + bsv[nn];
          float sig = 1.0f / (1.0f + __expf(-xf));
          float sp  = (xs > 20.0f) ? xs : __logf(1.0f + __expf(xs));
          int dn = (a == 0) ? drow0[i] : drow1[i];
          atomicAdd(&agg[(size_t)dn * HIDDEN + n], sig * sp);
        }
      }
    }
    __syncthreads();  // protect staging buffer (WAR) before next group
  }
}

// -------- pooling + final linear --------
__global__ void zero_kernel(float* __restrict__ sums, float* __restrict__ counts) {
  int tid = blockIdx.x * blockDim.x + threadIdx.x;
  if (tid < N_GRAPHS * HIDDEN) sums[tid] = 0.0f;
  else if (tid < N_GRAPHS * HIDDEN + N_GRAPHS) counts[tid - N_GRAPHS * HIDDEN] = 0.0f;
}

__global__ void pool_kernel(const float* __restrict__ h, const int* __restrict__ batch,
                            float* __restrict__ sums, float* __restrict__ counts) {
  int tid = blockIdx.x * blockDim.x + threadIdx.x;
  if (tid >= N_NODES * HIDDEN) return;
  int n = tid >> 7, c = tid & 127;
  int g = batch[n];
  atomicAdd(&sums[(size_t)g * HIDDEN + c], h[tid]);
  if (c == 0) atomicAdd(&counts[g], 1.0f);
}

__global__ void fin_kernel(const float* __restrict__ sums, const float* __restrict__ counts,
                           const float* __restrict__ Wlin, const float* __restrict__ blin,
                           float* __restrict__ out) {
  int tid = blockIdx.x * blockDim.x + threadIdx.x;
  if (tid >= N_GRAPHS * HIDDEN) return;
  int g = tid >> 7, j = tid & 127;
  float inv = 1.0f / fmaxf(counts[g], 1.0f);
  float acc = blin[j];
  for (int c = 0; c < HIDDEN; ++c)
    acc += sums[(size_t)g * HIDDEN + c] * inv * Wlin[(size_t)c * HIDDEN + j];
  out[tid] = acc;
}

extern "C" void kernel_launch(void* const* d_in, const int* in_sizes, int n_in,
                              void* d_out, int out_size, void* d_ws, size_t ws_size,
                              hipStream_t stream) {
  (void)in_sizes; (void)n_in; (void)out_size; (void)ws_size;
  const int*   x    = (const int*)d_in[0];
  const int*   esrc = (const int*)d_in[1];
  const int*   edst = esrc + N_EDGES;
  const float* ea   = (const float*)d_in[2];
  const int*   batch= (const int*)d_in[3];
  const float* emb  = (const float*)d_in[4];
  const float* Wf[3] = {(const float*)d_in[5],  (const float*)d_in[9],  (const float*)d_in[13]};
  const float* bf[3] = {(const float*)d_in[6],  (const float*)d_in[10], (const float*)d_in[14]};
  const float* Ws[3] = {(const float*)d_in[7],  (const float*)d_in[11], (const float*)d_in[15]};
  const float* bs[3] = {(const float*)d_in[8],  (const float*)d_in[12], (const float*)d_in[16]};
  const float* Wlin = (const float*)d_in[17];
  const float* blin = (const float*)d_in[18];
  float* out = (float*)d_out;

  // workspace layout (16B-aligned offsets); ~116 MB total
  char* ws = (char*)d_ws;
  float*          h      = (float*)(ws + 0);           // 25,600,000 B  f32 node feats
  float*          agg    = (float*)(ws + 25600000);    // 25,600,000 B  residual+scatter
  float*          sums   = (float*)(ws + 51200000);    //    256,000 B
  float*          counts = (float*)(ws + 51456000);    //      2,000 B (+pad)
  unsigned short* wt     = (unsigned short*)(ws + 51458048); // 442,368 B bf16 W^T x6
  unsigned short* hbf    = (unsigned short*)(ws + 51900416); // 12,800,000 B bf16 h
  unsigned short* eab    = (unsigned short*)(ws + 64700416); // 51,200,000 B bf16 edge_attr

  const int NH = N_NODES * HIDDEN;

  wconv_kernel<<<(6 * ZDIM * HIDDEN + 255) / 256, 256, 0, stream>>>(
      Wf[0], Ws[0], Wf[1], Ws[1], Wf[2], Ws[2], wt);
  eaconv_kernel<<<(N_EDGES * NUM_RBF + 255) / 256, 256, 0, stream>>>(ea, eab);
  embed_kernel<<<(NH + 255) / 256, 256, 0, stream>>>(x, emb, h, hbf);

  const size_t ldsBytes = (size_t)(2 * HIDDEN * ZPAD + 128 * ZPAD) * 2 + 128 * 4;
  for (int l = 0; l < 3; ++l) {
    copy4_kernel<<<(NH / 4 + 255) / 256, 256, 0, stream>>>(
        (const float4*)h, (float4*)agg, NH / 4);            // agg = h (residual)
    cg_msg_kernel<<<512, 256, ldsBytes, stream>>>(
        hbf, esrc, edst, eab,
        wt + (size_t)(2 * l) * ZDIM * HIDDEN,
        wt + (size_t)(2 * l + 1) * ZDIM * HIDDEN,
        bf[l], bs[l], agg);                                  // agg += messages
    relu_kernel<<<(NH + 255) / 256, 256, 0, stream>>>(agg, h, hbf, NH);
  }

  zero_kernel<<<(N_GRAPHS * HIDDEN + N_GRAPHS + 255) / 256, 256, 0, stream>>>(sums, counts);
  pool_kernel<<<(NH + 255) / 256, 256, 0, stream>>>(h, batch, sums, counts);
  fin_kernel<<<(N_GRAPHS * HIDDEN + 255) / 256, 256, 0, stream>>>(sums, counts, Wlin, blin, out);
}
